// ReEig_66760971649504
// MI455X (gfx1250) — compile-verified, matrix operand order checked by reading, not run on hardware
//
#include <hip/hip_runtime.h>
#include <hip/hip_bf16.h>
#include <math.h>

// ---------------------------------------------------------------------------
// ReEig(X) = U max(diag(w),eps) U^T  for SPD X, rewritten GEMM-only:
//   ReEig(X) = 0.5*(X + eps*I) + 0.5*(X - eps*I) * sign(X - eps*I)
// sign() via Newton-Schulz:  Y <- 1.5*Y - 0.5*Y^3   (all 4096^3 GEMMs)
// GEMMs run on v_wmma_f32_16x16x32_bf16 (bf16 in, f32 accumulate), with
// double-buffered LDS staging (one workgroup barrier per 64-wide K stage).
// ---------------------------------------------------------------------------

typedef __bf16 bf16;
typedef __attribute__((ext_vector_type(16))) __bf16 v16bf;
typedef __attribute__((ext_vector_type(8)))  __bf16 bf16x8;
typedef __attribute__((ext_vector_type(8)))  float  v8f;

#define NMAT 4096
#define EPSV 1e-4f
#define NS_ITERS 25
#define KSTEP 64
#define LSTRIDE 72          // 64 K-elems + 8 pad; 144B = 9*16B (aligned, staggered)

// ---------------- Frobenius norm (two-stage deterministic reduction) -------
__global__ void frob_partial_kernel(const float* __restrict__ X,
                                    float* __restrict__ partial) {
    __shared__ float s[256];
    const int b = blockIdx.x;                 // one row per block
    const float* row = X + (size_t)b * NMAT;
    float acc = 0.f;
    for (int i = threadIdx.x; i < NMAT; i += 256) {
        float v = row[i];
        acc += v * v;
    }
    s[threadIdx.x] = acc;
    __syncthreads();
    for (int off = 128; off > 0; off >>= 1) {
        if (threadIdx.x < off) s[threadIdx.x] += s[threadIdx.x + off];
        __syncthreads();
    }
    if (threadIdx.x == 0) partial[b] = s[0];
}

__global__ void frob_final_kernel(const float* __restrict__ partial,
                                  float* __restrict__ cnorm) {
    __shared__ float s[256];
    float acc = 0.f;
    for (int i = threadIdx.x; i < NMAT; i += 256) acc += partial[i];
    s[threadIdx.x] = acc;
    __syncthreads();
    for (int off = 128; off > 0; off >>= 1) {
        if (threadIdx.x < off) s[threadIdx.x] += s[threadIdx.x + off];
        __syncthreads();
    }
    if (threadIdx.x == 0) cnorm[0] = sqrtf(s[0]);
}

// --------------- M = X - eps*I (bf16),  Y0 = M / ||X||_F (bf16) ------------
__global__ void prep_kernel(const float* __restrict__ X,
                            bf16* __restrict__ Mb, bf16* __restrict__ Yb,
                            const float* __restrict__ cnorm) {
    const float inv = 1.0f / cnorm[0];
    size_t i = (size_t)blockIdx.x * 256 + threadIdx.x;
    size_t r = i / NMAT, c = i % NMAT;
    float m = X[i] - ((r == c) ? EPSV : 0.f);
    Mb[i] = (bf16)m;
    Yb[i] = (bf16)(m * inv);
}

// --------------------------- WMMA GEMM core --------------------------------
// mode 0: Cb = A*B                         (Z = Y*Y)
// mode 1: Cb = 1.5*Yold - 0.5*(A*B)        (NS update, A=Y, B=Z)
// mode 2: Cf = 0.5*(X + eps*I + A*B)       (final, A=M, B=sign)
// All operands are symmetric, so B-tiles are fetched row-wise (B^T cols).
__global__ __launch_bounds__(256, 2) void gemm_ns_kernel(
    const bf16* __restrict__ A, const bf16* __restrict__ B,
    const bf16* __restrict__ Yold, const float* __restrict__ X,
    bf16* __restrict__ Cb, float* __restrict__ Cf, int mode)
{
    // Double-buffered 128x64 bf16 tiles of A and B^T.
    __shared__ bf16 As[2][128 * LSTRIDE];
    __shared__ bf16 Bs[2][128 * LSTRIDE];

    const int tid  = threadIdx.x;
    const int lane = tid & 31;
    const int w    = tid >> 5;          // 8 waves
    const int wm   = w & 3;             // 4 along M
    const int wn   = w >> 2;            // 2 along N
    const int bm   = blockIdx.x, bn = blockIdx.y;
    const int rowA0 = bm * 128, rowB0 = bn * 128;
    const int halfsel = (lane >= 16) ? 8 : 0;   // K-half select per ISA layout
    const int lm = lane & 15;

    // Per-stage staging: 4 b128 chunks per tile per thread (128x64 bf16 tile).
    const int ldRow = tid >> 3;          // chunk row base for ch=0 (tid/8)
    const int ldCol = (tid & 7) * 8;     // chunk column (elements)

    v8f acc[2][4];
    const v8f vzero = {0.f, 0.f, 0.f, 0.f, 0.f, 0.f, 0.f, 0.f};
#pragma unroll
    for (int i = 0; i < 2; ++i)
#pragma unroll
        for (int j = 0; j < 4; ++j) acc[i][j] = vzero;

    bf16x8 ra[4], rb[4];

    // ---- prologue: stage K-slab 0 into buffer 0 ----
#pragma unroll
    for (int ch = 0; ch < 4; ++ch) {
        int r = ldRow + ch * 32;
        ra[ch] = *(const bf16x8*)(A + (size_t)(rowA0 + r) * NMAT + ldCol);
        rb[ch] = *(const bf16x8*)(B + (size_t)(rowB0 + r) * NMAT + ldCol);
    }
#pragma unroll
    for (int ch = 0; ch < 4; ++ch) {
        int r = ldRow + ch * 32;
        *(bf16x8*)&As[0][r * LSTRIDE + ldCol] = ra[ch];
        *(bf16x8*)&Bs[0][r * LSTRIDE + ldCol] = rb[ch];
    }
    __syncthreads();

    int cur = 0;
    for (int k0 = 0; k0 < NMAT; k0 += KSTEP) {
        const bool hasnext = (k0 + KSTEP < NMAT);
        // ---- issue global loads for the NEXT slab (latency hidden by WMMA) ----
        if (hasnext) {
#pragma unroll
            for (int ch = 0; ch < 4; ++ch) {
                int r = ldRow + ch * 32;
                ra[ch] = *(const bf16x8*)(A + (size_t)(rowA0 + r) * NMAT +
                                          k0 + KSTEP + ldCol);
                rb[ch] = *(const bf16x8*)(B + (size_t)(rowB0 + r) * NMAT +
                                          k0 + KSTEP + ldCol);
            }
        }

        // ---- compute: two 32-wide K slices from LDS[cur] ----
        const bf16* asb = &As[cur][0];
        const bf16* bsb = &Bs[cur][0];
#pragma unroll
        for (int ks = 0; ks < 2; ++ks) {
            const int koff = ks * 32;
            union Fr { v16bf v; bf16x8 h[2]; };
            Fr afr[2], bfr[4];
#pragma unroll
            for (int ti = 0; ti < 2; ++ti) {
                int m = wm * 32 + ti * 16 + lm;
                afr[ti].h[0] = *(const bf16x8*)&asb[m * LSTRIDE + koff + halfsel];
                afr[ti].h[1] = *(const bf16x8*)&asb[m * LSTRIDE + koff + 16 + halfsel];
            }
#pragma unroll
            for (int tj = 0; tj < 4; ++tj) {
                int nn = wn * 64 + tj * 16 + lm;
                bfr[tj].h[0] = *(const bf16x8*)&bsb[nn * LSTRIDE + koff + halfsel];
                bfr[tj].h[1] = *(const bf16x8*)&bsb[nn * LSTRIDE + koff + 16 + halfsel];
            }
#pragma unroll
            for (int ti = 0; ti < 2; ++ti)
#pragma unroll
                for (int tj = 0; tj < 4; ++tj)
                    acc[ti][tj] = __builtin_amdgcn_wmma_f32_16x16x32_bf16(
                        false, afr[ti].v, false, bfr[tj].v,
                        (short)0, acc[ti][tj], false, false);
        }

        // ---- drain staged regs into the other buffer; ONE barrier per stage ----
        if (hasnext) {
            const int nxt = cur ^ 1;
#pragma unroll
            for (int ch = 0; ch < 4; ++ch) {
                int r = ldRow + ch * 32;
                *(bf16x8*)&As[nxt][r * LSTRIDE + ldCol] = ra[ch];
                *(bf16x8*)&Bs[nxt][r * LSTRIDE + ldCol] = rb[ch];
            }
        }
        __syncthreads();
        cur ^= 1;
    }

    // Epilogue: C/D layout = VGPR j -> M=j (+8 for lanes 16..31), N=lane%16.
#pragma unroll
    for (int ti = 0; ti < 2; ++ti) {
#pragma unroll
        for (int tj = 0; tj < 4; ++tj) {
            const int m0 = bm * 128 + wm * 32 + ti * 16;
            const int n0 = bn * 128 + wn * 64 + tj * 16;
            const int ncol  = n0 + lm;
            const int mbase = m0 + ((lane >= 16) ? 8 : 0);
#pragma unroll
            for (int j = 0; j < 8; ++j) {
                const int m = mbase + j;
                const float a = acc[ti][tj][j];
                const size_t idx = (size_t)m * NMAT + ncol;
                if (mode == 0) {
                    Cb[idx] = (bf16)a;
                } else if (mode == 1) {
                    float y = (float)Yold[idx];
                    Cb[idx] = (bf16)(1.5f * y - 0.5f * a);
                } else {
                    float x = X[idx];
                    Cf[idx] = 0.5f * (x + ((m == ncol) ? EPSV : 0.f) + a);
                }
            }
        }
    }
}

// ---------------------------------------------------------------------------
extern "C" void kernel_launch(void* const* d_in, const int* in_sizes, int n_in,
                              void* d_out, int out_size, void* d_ws, size_t ws_size,
                              hipStream_t stream) {
    (void)in_sizes; (void)n_in; (void)out_size; (void)ws_size;
    const float* X = (const float*)d_in[0];
    float* out = (float*)d_out;

    const size_t NN = (size_t)NMAT * NMAT;
    bf16*  Mb      = (bf16*)d_ws;          // 32 MB: bf16(X - eps I)
    bf16*  Ya      = Mb + NN;              // 32 MB: NS iterate (ping)
    bf16*  Yb      = Ya + NN;              // 32 MB: NS iterate (pong)
    bf16*  Z       = Yb + NN;              // 32 MB: Y*Y
    float* partial = (float*)(Z + NN);     // 16 KB
    float* cnorm   = partial + NMAT;       // 4 B

    frob_partial_kernel<<<NMAT, 256, 0, stream>>>(X, partial);
    frob_final_kernel<<<1, 256, 0, stream>>>(partial, cnorm);
    prep_kernel<<<(unsigned)(NN / 256), 256, 0, stream>>>(X, Mb, Ya, cnorm);

    dim3 grid(NMAT / 128, NMAT / 128), block(256);
    bf16* Ycur = Ya;
    bf16* Ynxt = Yb;
    for (int it = 0; it < NS_ITERS; ++it) {
        // Z = Y*Y
        gemm_ns_kernel<<<grid, block, 0, stream>>>(
            Ycur, Ycur, nullptr, nullptr, Z, nullptr, 0);
        // Ynext = 1.5*Y - 0.5*Y*Z
        gemm_ns_kernel<<<grid, block, 0, stream>>>(
            Ycur, Z, Ycur, nullptr, Ynxt, nullptr, 1);
        bf16* t = Ycur; Ycur = Ynxt; Ynxt = t;
    }
    // out = 0.5*(X + eps*I + (X - eps*I)*sign(X - eps*I))
    gemm_ns_kernel<<<grid, block, 0, stream>>>(
        Mb, Ycur, nullptr, X, nullptr, out, 2);
}
